// LayoutLMv3Encoder_19473381720545
// MI455X (gfx1250) — matrix operand order repaired and checked by reading, not compiled, not run on hardware
//
#include <hip/hip_runtime.h>
#include <hip/hip_bf16.h>

typedef _Float16 half_t;
typedef __attribute__((ext_vector_type(16))) _Float16 v16h;
typedef __attribute__((ext_vector_type(8)))  _Float16 v8h;
typedef __attribute__((ext_vector_type(8)))  float    v8f;
typedef __attribute__((ext_vector_type(4)))  int      v4i;

union V16 { v16h v; v8h h[2]; };

static constexpr int cB  = 4;
static constexpr int cS  = 512;
static constexpr int cH  = 768;
static constexpr int cNH = 12;
static constexpr int cDH = 64;
static constexpr int cFF = 3072;
static constexpr int cL  = 12;
static constexpr int cM  = cB * cS;      // 2048 rows of activations
static constexpr float cSCALE = 0.125f;  // 1/sqrt(64)

#define GLOBAL_AS __attribute__((address_space(1)))
#define LDS_AS    __attribute__((address_space(3)))

// --- CDNA5 async global->LDS copy (ASYNCcnt-tracked), guarded for portability ---
#if defined(__has_builtin)
#if __has_builtin(__builtin_amdgcn_global_load_async_to_lds_b128)
#define USE_ASYNC_LDS 1
#endif
#endif

#ifdef USE_ASYNC_LDS
__device__ __forceinline__ void async_b128(const half_t* g, half_t* l) {
    __builtin_amdgcn_global_load_async_to_lds_b128(
        (GLOBAL_AS v4i*)(void*)g, (LDS_AS v4i*)(void*)l, 0, 0);
}
__device__ __forceinline__ void wait_async0() {
#if __has_builtin(__builtin_amdgcn_s_wait_asynccnt)
    __builtin_amdgcn_s_wait_asynccnt(0);
#else
    asm volatile("s_wait_asynccnt 0" ::: "memory");
#endif
}
#endif

// ---------------------------------------------------------------------------
// Relative-position bias:  bias[b,h,i,j] = (relw[h,b1]+relx[h,bx]+rely[h,by])*scale + mask[b,j]
// ---------------------------------------------------------------------------
__device__ __forceinline__ int bucket_fn(int rel, int nb, int maxd) {
    int half_ = nb / 2;
    int ret = (rel > 0) ? half_ : 0;
    int n = rel < 0 ? -rel : rel;
    int max_exact = half_ / 2;
    float t = logf(fmaxf((float)n, 1.0f) / (float)max_exact) /
              logf((float)maxd / (float)max_exact) * (float)(half_ - max_exact);
    int val_large = max_exact + (int)t;           // trunc toward zero, matches astype(int32)
    val_large = min(val_large, half_ - 1);
    return ret + ((n < max_exact) ? n : val_large);
}

__global__ __launch_bounds__(256)
void bias_kernel(const int* __restrict__ pos, const int* __restrict__ bbox,
                 const float* __restrict__ mask,
                 const float* __restrict__ relw, const float* __restrict__ relx,
                 const float* __restrict__ rely, float* __restrict__ bias)
{
    size_t idx = (size_t)blockIdx.x * 256 + threadIdx.x;
    if (idx >= (size_t)cB * cS * cS) return;
    int j = idx % cS; size_t r = idx / cS;
    int i = r % cS;  int b = r / cS;
    int b1 = bucket_fn(pos[b*cS + j] - pos[b*cS + i], 32, 128);
    int cxi = bbox[((size_t)b*cS + i)*4 + 0], cxj = bbox[((size_t)b*cS + j)*4 + 0];
    int cyi = bbox[((size_t)b*cS + i)*4 + 3], cyj = bbox[((size_t)b*cS + j)*4 + 3];
    int bx = bucket_fn(cxj - cxi, 64, 256);
    int by = bucket_fn(cyj - cyi, 64, 256);
    float mk = mask[b*cS + j];
#pragma unroll
    for (int h = 0; h < cNH; ++h) {
        float v = (relw[h*32 + b1] + relx[h*64 + bx] + rely[h*64 + by]) * cSCALE + mk;
        bias[(((size_t)b*cNH + h)*cS + i)*cS + j] = v;
    }
}

// ---------------------------------------------------------------------------
// f32 -> f16 convert (activations)
// ---------------------------------------------------------------------------
__global__ __launch_bounds__(256)
void cvt_f32_f16(const float* __restrict__ x, half_t* __restrict__ y, int n)
{
    int i = blockIdx.x * 256 + threadIdx.x;
    if (i < n) y[i] = (half_t)x[i];
}

// W[k*N+n] (f32) -> WT[n*K+k] (f16): per-lane K-contiguous B operand for WMMA
__global__ __launch_bounds__(256)
void transpose_cvt(const float* __restrict__ W, half_t* __restrict__ WT, int K, int N)
{
    size_t idx = (size_t)blockIdx.x * 256 + threadIdx.x;
    if (idx >= (size_t)K * N) return;
    int k = idx % K; int n = idx / K;
    WT[idx] = (half_t)W[(size_t)k * N + n];
}

// ---------------------------------------------------------------------------
// WMMA GEMM: C[M,N] = A[M,K](f16) * B (BT = [N][K] f16) + bias[n], templated epilogue
// EPI 0: f16 out, *scale.  EPI 1: f32 out.  EPI 2: exact-GELU, f16 out.
// Block tile 128x128, 8 waves (4x2), wave tile 32x64, K-step 32.
// Double-buffered LDS; async global->LDS (ASYNCcnt) when available.
// ---------------------------------------------------------------------------
template <int EPI>
__global__ __launch_bounds__(256)
void gemm_f16(const half_t* __restrict__ A, const half_t* __restrict__ BT,
              const float* __restrict__ bias, void* __restrict__ Out,
              int M, int N, int K, float scale)
{
    __shared__ half_t As[2][128 * 40];
    __shared__ half_t Bs[2][128 * 40];

    const int bn = blockIdx.x, bm = blockIdx.y;
    const int t = threadIdx.x;
    const int wid = t >> 5, lane = t & 31;
    const int wm = wid >> 1, wn = wid & 1;
    const int li = lane & 15, hi = lane >> 4;

    v8f acc[2][4];
#pragma unroll
    for (int i = 0; i < 2; ++i)
#pragma unroll
        for (int j = 0; j < 4; ++j) acc[i][j] = {};

    const int ldr  = t >> 1;          // 0..127: tile row this thread stages
    const int lseg = (t & 1) * 16;    // 0 / 16: half-row segment

    const half_t* gA = A  + (size_t)(bm*128 + ldr) * K + lseg;
    const half_t* gB = BT + (size_t)(bn*128 + ldr) * K + lseg;

    auto stage = [&](int kt, int buf) {
        const half_t* a = gA + kt * 32;
        const half_t* b = gB + kt * 32;
        half_t* la = &As[buf][ldr*40 + lseg];
        half_t* lb = &Bs[buf][ldr*40 + lseg];
#ifdef USE_ASYNC_LDS
        async_b128(a,     la);
        async_b128(a + 8, la + 8);
        async_b128(b,     lb);
        async_b128(b + 8, lb + 8);
#else
        *(v8h*)la       = *(const v8h*)a;
        *(v8h*)(la + 8) = *(const v8h*)(a + 8);
        *(v8h*)lb       = *(const v8h*)b;
        *(v8h*)(lb + 8) = *(const v8h*)(b + 8);
#endif
    };

    const int nk = K / 32;
    stage(0, 0);
#ifdef USE_ASYNC_LDS
    wait_async0();
#endif
    __syncthreads();

    for (int kt = 0; kt < nk; ++kt) {
        const int cur = kt & 1;
        if (kt + 1 < nk) stage(kt + 1, cur ^ 1);   // overlap with WMMAs below

        v16h af[2], bfr[4];
#pragma unroll
        for (int i = 0; i < 2; ++i) {   // A frag: lane=row, halves {hi*8..}, {16+hi*8..}
            const half_t* p = &As[cur][(wm*32 + i*16 + li) * 40];
            V16 u; u.h[0] = *(const v8h*)(p + hi*8);
                   u.h[1] = *(const v8h*)(p + 16 + hi*8);
            af[i] = u.v;
        }
#pragma unroll
        for (int j = 0; j < 4; ++j) {   // B frag: lane=col, halves {hi*8..}, {16+hi*8..}
            const half_t* p = &Bs[cur][(wn*64 + j*16 + li) * 40];
            V16 u; u.h[0] = *(const v8h*)(p + hi*8);
                   u.h[1] = *(const v8h*)(p + 16 + hi*8);
            bfr[j] = u.v;
        }
#pragma unroll
        for (int i = 0; i < 2; ++i)
#pragma unroll
            for (int j = 0; j < 4; ++j)
                acc[i][j] = __builtin_amdgcn_wmma_f32_16x16x32_f16(
                    false, af[i], false, bfr[j], (short)0, acc[i][j], false, false);

#ifdef USE_ASYNC_LDS
        wait_async0();       // my async writes to the other buffer have landed
#endif
        __syncthreads();     // everyone's have
    }

#pragma unroll
    for (int i = 0; i < 2; ++i)
#pragma unroll
        for (int j = 0; j < 4; ++j) {
            int col = bn*128 + wn*64 + j*16 + li;
            float bv = bias[col];
#pragma unroll
            for (int g = 0; g < 8; ++g) {
                int row = bm*128 + wm*32 + i*16 + g + 8*hi;
                float v = acc[i][j][g] + bv;
                if (EPI == 2) v = 0.5f * v * (1.0f + erff(v * 0.70710678118654752f));
                if (EPI == 1) ((float*) Out)[(size_t)row * N + col] = v;
                else          ((half_t*)Out)[(size_t)row * N + col] = (half_t)(v * scale);
            }
        }
}

// ---------------------------------------------------------------------------
// Flash-style attention: block = (b, head, 128 q-rows); 8 waves x 16 rows.
// Q pre-scaled by 1/sqrt(DH); bias tensor already contains mask + scaled rel bias.
// ---------------------------------------------------------------------------
__global__ __launch_bounds__(256)
void attn_kernel(const half_t* __restrict__ q, const half_t* __restrict__ k,
                 const half_t* __restrict__ v, const float* __restrict__ bias,
                 half_t* __restrict__ ctxOut)
{
    __shared__ half_t Vt[64 * 40];        // V tile transposed: [d][key]
    __shared__ half_t Pl[8 * 16 * 40];    // per-wave probabilities: [row][key]

    int blk = blockIdx.x;
    int qb = blk % (cS / 128); int rest = blk / (cS / 128);
    int head = rest % cNH;     int b = rest / cNH;

    const int t = threadIdx.x, wid = t >> 5, lane = t & 31;
    const int li = lane & 15, hi = lane >> 4;
    const int qrow0 = qb * 128 + wid * 16;

    // Q fragments (16x64 -> 2 frags of 16x32), straight from global
    v16h qf[2];
    const half_t* qbase = q + ((size_t)(b*cS + qrow0)) * cH + head * cDH;
#pragma unroll
    for (int i = 0; i < 2; ++i) {
        const half_t* p = qbase + (size_t)li * cH + i * 32;
        V16 u; u.h[0] = *(const v8h*)(p + hi*8);
               u.h[1] = *(const v8h*)(p + 16 + hi*8);
        qf[i] = u.v;
    }

    v8f ctx[4]; float mrow[8], lrow[8];
#pragma unroll
    for (int j = 0; j < 4; ++j) ctx[j] = {};
#pragma unroll
    for (int g = 0; g < 8; ++g) { mrow[g] = -3.0e38f; lrow[g] = 0.0f; }

    const float* biasRow = bias + (((size_t)(b*cNH + head))*cS + qrow0) * cS;

    for (int kt = 0; kt < cS; kt += 32) {
        // prefetch next key-tile's bias rows (global_prefetch_b8) behind the WMMAs
        if (kt + 32 < cS) {
#pragma unroll
            for (int g = 0; g < 8; ++g)
                __builtin_prefetch(biasRow + (size_t)(g + 8*hi) * cS + kt + 32 + li, 0, 2);
        }
        __syncthreads();
        { // stage V[32 keys][64 d] transposed -> Vt[d][key]
            int key = t >> 3, dg = (t & 7) * 8;
            const half_t* vp = v + ((size_t)(b*cS + kt + key)) * cH + head * cDH + dg;
            v8h vv = *(const v8h*)vp;
#pragma unroll
            for (int e = 0; e < 8; ++e) Vt[(dg + e) * 40 + key] = vv[e];
        }
        __syncthreads();

        // scores: two 16x16 tiles over 32 keys
        v8f sc[2];
#pragma unroll
        for (int n = 0; n < 2; ++n) {
            v8f a = {};
#pragma unroll
            for (int kk = 0; kk < 2; ++kk) {
                const half_t* kp = k + ((size_t)(b*cS + kt + n*16 + li)) * cH
                                     + head * cDH + kk * 32 + hi * 8;
                V16 u; u.h[0] = *(const v8h*)kp; u.h[1] = *(const v8h*)(kp + 16);
                a = __builtin_amdgcn_wmma_f32_16x16x32_f16(
                        false, qf[kk], false, u.v, (short)0, a, false, false);
            }
            sc[n] = a;
        }

        // bias add + online softmax update (row g+8*hi lives across 16 lanes)
#pragma unroll
        for (int g = 0; g < 8; ++g) {
            int r = g + 8 * hi;
            float b0 = biasRow[(size_t)r * cS + kt + li];
            float b1 = biasRow[(size_t)r * cS + kt + 16 + li];
            float s0 = sc[0][g] + b0, s1 = sc[1][g] + b1;
            float mx = fmaxf(s0, s1);
#pragma unroll
            for (int d = 8; d >= 1; d >>= 1) mx = fmaxf(mx, __shfl_xor(mx, d, 16));
            float nm = fmaxf(mrow[g], mx);
            float fac = __expf(mrow[g] - nm);
            mrow[g] = nm;
            float p0 = __expf(s0 - nm), p1 = __expf(s1 - nm);
            float rs = p0 + p1;
#pragma unroll
            for (int d = 8; d >= 1; d >>= 1) rs += __shfl_xor(rs, d, 16);
            lrow[g] = lrow[g] * fac + rs;
#pragma unroll
            for (int j = 0; j < 4; ++j) ctx[j][g] *= fac;
            Pl[(wid*16 + r) * 40 + li]      = (half_t)p0;
            Pl[(wid*16 + r) * 40 + 16 + li] = (half_t)p1;
        }

        // re-fragment probabilities as A (LDS ops in-order within wave)
        V16 up;
        { const half_t* p = &Pl[(wid*16 + li) * 40];
          up.h[0] = *(const v8h*)(p + hi*8);
          up.h[1] = *(const v8h*)(p + 16 + hi*8); }

#pragma unroll
        for (int j = 0; j < 4; ++j) {   // PV: B = V (32 keys x 16 d), per-lane K-contig from Vt
            const half_t* vp = &Vt[(j*16 + li) * 40 + hi * 8];
            V16 u; u.h[0] = *(const v8h*)vp; u.h[1] = *(const v8h*)(vp + 16);
            ctx[j] = __builtin_amdgcn_wmma_f32_16x16x32_f16(
                         false, up.v, false, u.v, (short)0, ctx[j], false, false);
        }
    }

    half_t* obase = ctxOut + ((size_t)(b*cS + qrow0)) * cH + head * cDH;
#pragma unroll
    for (int j = 0; j < 4; ++j)
#pragma unroll
        for (int g = 0; g < 8; ++g) {
            int r = g + 8 * hi;
            obase[(size_t)r * cH + j*16 + li] = (half_t)(ctx[j][g] / lrow[g]);
        }
}

// ---------------------------------------------------------------------------
// Residual + LayerNorm: y = LN(x + res)*g + b; writes f32 and f16 copies.
// One block per row of 768.
// ---------------------------------------------------------------------------
__global__ __launch_bounds__(256)
void ln_kernel(const float* __restrict__ x, const float* __restrict__ res,
               const float* __restrict__ g, const float* __restrict__ be,
               float* __restrict__ out32, half_t* __restrict__ out16)
{
    __shared__ float sh1[256], sh2[256];
    const int row = blockIdx.x, t = threadIdx.x;
    const float* xr = x   + (size_t)row * cH;
    const float* rr = res + (size_t)row * cH;
    float vals[3], s = 0.f, ss = 0.f;
#pragma unroll
    for (int i = 0; i < 3; ++i) {
        int c = t + i * 256;
        float vv = xr[c] + rr[c];
        vals[i] = vv; s += vv; ss += vv * vv;
    }
    sh1[t] = s; sh2[t] = ss; __syncthreads();
    for (int off = 128; off > 0; off >>= 1) {
        if (t < off) { sh1[t] += sh1[t + off]; sh2[t] += sh2[t + off]; }
        __syncthreads();
    }
    float mean = sh1[0] * (1.0f / cH);
    float var  = sh2[0] * (1.0f / cH) - mean * mean;
    float inv  = rsqrtf(var + 1e-5f);
#pragma unroll
    for (int i = 0; i < 3; ++i) {
        int c = t + i * 256;
        float y = (vals[i] - mean) * inv * g[c] + be[c];
        out32[(size_t)row * cH + c] = y;
        out16[(size_t)row * cH + c] = (half_t)y;
    }
}

// ---------------------------------------------------------------------------
// Host orchestration
// ---------------------------------------------------------------------------
extern "C" void kernel_launch(void* const* d_in, const int* in_sizes, int n_in,
                              void* d_out, int out_size, void* d_ws, size_t ws_size,
                              hipStream_t stream)
{
    (void)in_sizes; (void)n_in; (void)out_size; (void)ws_size;
    const float* hidden = (const float*)d_in[0];
    const float* mask   = (const float*)d_in[1];
    const int*   pos    = (const int*)  d_in[2];
    const int*   bbox   = (const int*)  d_in[3];
    const float* Wq = (const float*)d_in[4];  const float* bq = (const float*)d_in[5];
    const float* Wk = (const float*)d_in[6];  const float* bk = (const float*)d_in[7];
    const float* Wv = (const float*)d_in[8];  const float* bv = (const float*)d_in[9];
    const float* Wo = (const float*)d_in[10]; const float* bo = (const float*)d_in[11];
    const float* l1g = (const float*)d_in[12]; const float* l1b = (const float*)d_in[13];
    const float* Wi = (const float*)d_in[14]; const float* bi = (const float*)d_in[15];
    const float* Wf = (const float*)d_in[16]; const float* bf = (const float*)d_in[17];
    const float* l2g = (const float*)d_in[18]; const float* l2b = (const float*)d_in[19];
    const float* relw = (const float*)d_in[20];
    const float* relx = (const float*)d_in[21];
    const float* rely = (const float*)d_in[22];

    char* wp = (char*)d_ws;
    auto alloc = [&](size_t bytes) -> void* {
        void* r = (void*)wp; wp += (bytes + 255) & ~(size_t)255; return r;
    };
    float*  biasT   = (float*) alloc((size_t)cB*cNH*cS*cS*4);   // 50.3 MB
    half_t* WqT     = (half_t*)alloc((size_t)cH*cH*2);
    half_t* WkT     = (half_t*)alloc((size_t)cH*cH*2);
    half_t* WvT     = (half_t*)alloc((size_t)cH*cH*2);
    half_t* WoT     = (half_t*)alloc((size_t)cH*cH*2);
    half_t* WiT     = (half_t*)alloc((size_t)cH*cFF*2);
    half_t* WfT     = (half_t*)alloc((size_t)cFF*cH*2);
    half_t* h16     = (half_t*)alloc((size_t)cM*cH*2);
    half_t* q16     = (half_t*)alloc((size_t)cM*cH*2);
    half_t* k16     = (half_t*)alloc((size_t)cM*cH*2);
    half_t* v16     = (half_t*)alloc((size_t)cM*cH*2);
    half_t* ctx16   = (half_t*)alloc((size_t)cM*cH*2);
    half_t* attn16  = (half_t*)alloc((size_t)cM*cH*2);
    half_t* inter16 = (half_t*)alloc((size_t)cM*cFF*2);
    float*  tmp32   = (float*) alloc((size_t)cM*cH*4);
    float*  attn32  = (float*) alloc((size_t)cM*cH*4);
    float*  h32a    = (float*) alloc((size_t)cM*cH*4);
    float*  h32b    = (float*) alloc((size_t)cM*cH*4);

    bias_kernel<<<(cB*cS*cS)/256, 256, 0, stream>>>(pos, bbox, mask, relw, relx, rely, biasT);
    cvt_f32_f16<<<(cM*cH)/256, 256, 0, stream>>>(hidden, h16, cM*cH);

    const dim3 g768(cH/128, cM/128), g3072(cFF/128, cM/128);
    const int grHH = (cH*cH)/256, grHF = (cH*cFF)/256;

    const float* hres = hidden;             // residual for LN1 (f32)
    float* hbuf[2] = { h32a, h32b };

    for (int l = 0; l < cL; ++l) {
        transpose_cvt<<<grHH, 256, 0, stream>>>(Wq + (size_t)l*cH*cH,  WqT, cH, cH);
        transpose_cvt<<<grHH, 256, 0, stream>>>(Wk + (size_t)l*cH*cH,  WkT, cH, cH);
        transpose_cvt<<<grHH, 256, 0, stream>>>(Wv + (size_t)l*cH*cH,  WvT, cH, cH);
        transpose_cvt<<<grHH, 256, 0, stream>>>(Wo + (size_t)l*cH*cH,  WoT, cH, cH);
        transpose_cvt<<<grHF, 256, 0, stream>>>(Wi + (size_t)l*cH*cFF, WiT, cH, cFF);
        transpose_cvt<<<grHF, 256, 0, stream>>>(Wf + (size_t)l*cFF*cH, WfT, cFF, cH);

        gemm_f16<0><<<g768, 256, 0, stream>>>(h16, WqT, bq + l*cH, q16, cM, cH, cH, cSCALE);
        gemm_f16<0><<<g768, 256, 0, stream>>>(h16, WkT, bk + l*cH, k16, cM, cH, cH, 1.0f);
        gemm_f16<0><<<g768, 256, 0, stream>>>(h16, WvT, bv + l*cH, v16, cM, cH, cH, 1.0f);

        attn_kernel<<<cB*cNH*(cS/128), 256, 0, stream>>>(q16, k16, v16, biasT, ctx16);

        gemm_f16<1><<<g768, 256, 0, stream>>>(ctx16, WoT, bo + l*cH, tmp32, cM, cH, cH, 1.0f);
        ln_kernel<<<cM, 256, 0, stream>>>(tmp32, hres, l1g + l*cH, l1b + l*cH, attn32, attn16);

        gemm_f16<2><<<g3072, 256, 0, stream>>>(attn16, WiT, bi + l*cFF, inter16, cM, cFF, cH, 1.0f);
        gemm_f16<1><<<g768, 256, 0, stream>>>(inter16, WfT, bf + l*cH, tmp32, cM, cH, cFF, 1.0f);

        float* out32 = (l == cL-1) ? (float*)d_out : hbuf[l & 1];
        ln_kernel<<<cM, 256, 0, stream>>>(tmp32, attn32, l2g + l*cH, l2b + l*cH, out32, h16);
        hres = out32;
    }
}
